// SelfAttention_80702435492474
// MI455X (gfx1250) — matrix operand order repaired and checked by reading, not compiled, not run on hardware
//
#include <hip/hip_runtime.h>

// Self-attention (column-softmax variant), B=4, C=256, H=W=64 -> N=4096, d=64.
// Pipeline (all bf16 WMMA 16x16x32, f32 accumulate):
//   1) qkv_kernel x3 : Q,K,V = x^T @ W  (bf16 outputs in workspace)
//   2) stats_kernel  : per-COLUMN (query-axis) max and sum(exp) of S = Q K^T
//   3) attn_out      : recompute S tiles, P = exp(S-max)/sum, O += P @ V,
//                      out = gamma*O + x (transposed store back to BCHW)
// Double-buffered LDS staging so global->LDS copies overlap the WMMA stream.

#define B_ 4
#define C_ 256
#define N_ 4096
#define D_ 64

typedef __attribute__((ext_vector_type(16))) __bf16 v16bf;
typedef __attribute__((ext_vector_type(8)))  float  v8f;

union BF16Frag { unsigned int u[8]; v16bf v; };

__device__ __forceinline__ unsigned short f2bf(float f) {
  unsigned int u = __float_as_uint(f);
  u += 0x7fffu + ((u >> 16) & 1u);          // round-to-nearest-even
  return (unsigned short)(u >> 16);
}
__device__ __forceinline__ unsigned int pack2(unsigned short lo, unsigned short hi) {
  return (unsigned int)lo | ((unsigned int)hi << 16);
}

// A-matrix 16x32 bf16 fragment (ISA 7.12.2): lane<16: M=lane, VGPR j<4 K={2j,2j+1},
// j>=4 K={16+2(j-4),..}; lanes 16-31: M=lane-16, K offset +8. ku_base in uint units.
__device__ __forceinline__ v16bf frag_a(const unsigned int* p, int row, int stride_u,
                                        int ku_base, int lane) {
  BF16Frag t;
  const unsigned int* q = p + row * stride_u + ku_base + ((lane >= 16) ? 4 : 0);
#pragma unroll
  for (int j = 0; j < 8; ++j) t.u[j] = q[(j < 4) ? j : (j + 4)];
  return t.v;
}
// B-matrix 32x16 bf16 fragment: lane<16: N=lane, VGPR j<4 K={2j,2j+1},
// j>=4 K={16+2(j-4),..}; lanes 16-31: N=lane-16, K offset +8.  Staged as Bt[N][K] (K contiguous).
__device__ __forceinline__ v16bf frag_b(const unsigned int* p, int row, int stride_u,
                                        int ku_base, int lane) {
  BF16Frag t;
  const unsigned int* q = p + row * stride_u + ku_base + ((lane >= 16) ? 4 : 0);
#pragma unroll
  for (int j = 0; j < 8; ++j) t.u[j] = q[(j < 4) ? j : (j + 4)];
  return t.v;
}
__device__ __forceinline__ v8f wmma_bf16(v16bf a, v16bf b, v8f c) {
  return __builtin_amdgcn_wmma_f32_16x16x32_bf16(false, a, false, b, (short)0, c,
                                                 false, false);
}

// ---------------- Kernel 1: projection  out[n][d] = sum_c x[b][c][n] * W[c][d] ---
__global__ __launch_bounds__(256)
void qkv_kernel(const float* __restrict__ x, const float* __restrict__ W,
                unsigned short* __restrict__ outbf, int Dout) {
  __shared__ unsigned int wsT_u[64 * 129];                 // W^T chunk [64 d][256 c] bf16
  unsigned short* wsT_bf = (unsigned short*)wsT_u;         // row stride 258 bf16

  const int b = blockIdx.x >> 5, nt = blockIdx.x & 31;
  const int n0 = nt * 128;
  const int tid = threadIdx.x, w = tid >> 5, lane = tid & 31;
  const int hi = (lane >= 16) ? 1 : 0;
  const int nrow = n0 + 16 * w + (lane & 15);
  const float* xb = x + (size_t)b * C_ * N_;

  // A fragments (x^T rows) held in registers: 8 K-steps over C=256
  v16bf af[8];
#pragma unroll
  for (int kk = 0; kk < 8; ++kk) {
    BF16Frag t;
#pragma unroll
    for (int j = 0; j < 8; ++j) {
      int c0 = 32 * kk + ((j < 4) ? 2 * j : 16 + 2 * (j - 4)) + 8 * hi;
      float f0 = xb[(size_t)c0 * N_ + nrow];
      float f1 = xb[(size_t)(c0 + 1) * N_ + nrow];
      t.u[j] = pack2(f2bf(f0), f2bf(f1));
    }
    af[kk] = t.v;
  }

  const int ndc = Dout >> 6;
  for (int dc = 0; dc < ndc; ++dc) {
    // stage W^T chunk as bf16 (b128 global reads along d)
    for (int idx = tid; idx < 256 * 16; idx += 256) {
      int c = idx >> 4, d4 = idx & 15;
      float4 wv = *reinterpret_cast<const float4*>(&W[(size_t)c * Dout + dc * 64 + 4 * d4]);
      wsT_bf[(4 * d4 + 0) * 258 + c] = f2bf(wv.x);
      wsT_bf[(4 * d4 + 1) * 258 + c] = f2bf(wv.y);
      wsT_bf[(4 * d4 + 2) * 258 + c] = f2bf(wv.z);
      wsT_bf[(4 * d4 + 3) * 258 + c] = f2bf(wv.w);
    }
    __syncthreads();
#pragma unroll
    for (int t4 = 0; t4 < 4; ++t4) {
      v8f acc = {};
#pragma unroll
      for (int kk = 0; kk < 8; ++kk) {
        v16bf bfr = frag_b(wsT_u, 16 * t4 + (lane & 15), 129, 16 * kk, lane);
        acc = wmma_bf16(af[kk], bfr, acc);
      }
      int dcol = dc * 64 + 16 * t4 + (lane & 15);
#pragma unroll
      for (int j = 0; j < 8; ++j) {
        int m = j + 8 * hi;
        outbf[(size_t)(n0 + 16 * w + m) * Dout + dcol] = f2bf(acc[j]);
      }
    }
    __syncthreads();
  }
}

// ---------------- Kernel 2: column (query-axis) softmax stats of S = Q K^T -------
__global__ __launch_bounds__(256)
void stats_kernel(const unsigned short* __restrict__ Qbf,
                  const unsigned short* __restrict__ Kbf,
                  float* __restrict__ colmax, float* __restrict__ colsum) {
  __shared__ unsigned int qs_u[2][128 * 36];               // Q chunk [128 q][64 d], ping-pong
  __shared__ float cm_l[128];
  const int b = blockIdx.x >> 5, kt = blockIdx.x & 31;
  const int k0 = kt * 128;
  const int tid = threadIdx.x, w = tid >> 5, lane = tid & 31;
  const unsigned int* Qu = (const unsigned int*)(Qbf + (size_t)b * N_ * D_);
  const unsigned int* Ku = (const unsigned int*)(Kbf + (size_t)b * N_ * D_);
  const uint4* Qu4 = reinterpret_cast<const uint4*>(Qu);

  // Each wave owns 16 k-columns; B fragments (K rows of k) held in registers.
  v16bf kb[2];
#pragma unroll
  for (int kk = 0; kk < 2; ++kk)
    kb[kk] = frag_b(Ku, k0 + 16 * w + (lane & 15), 32, 16 * kk, lane);

  auto stage = [&](int qc, int buf) {
    for (int idx = tid; idx < 128 * 8; idx += 256) {       // b128 loads, aligned b128 LDS stores
      int r = idx >> 3, cu4 = idx & 7;
      uint4 val = Qu4[(size_t)(qc * 128 + r) * 8 + cu4];
      *reinterpret_cast<uint4*>(&qs_u[buf][r * 36 + 4 * cu4]) = val;
    }
  };

  float runmax = -3.0e38f;
  stage(0, 0);
  __syncthreads();
  for (int qc = 0; qc < 32; ++qc) {
    const int buf = qc & 1;
    if (qc + 1 < 32) stage(qc + 1, buf ^ 1);
#pragma unroll
    for (int qt2 = 0; qt2 < 8; ++qt2) {
      v8f s = {};
#pragma unroll
      for (int kk = 0; kk < 2; ++kk) {
        v16bf a = frag_a(&qs_u[buf][0], 16 * qt2 + (lane & 15), 36, 16 * kk, lane);
        s = wmma_bf16(a, kb[kk], s);
      }
#pragma unroll
      for (int j = 0; j < 8; ++j) runmax = fmaxf(runmax, s[j]);
    }
    __syncthreads();
  }
  runmax = fmaxf(runmax, __shfl_xor(runmax, 16, 32));      // lanes n,n+16 share a column
  if (lane < 16) cm_l[16 * w + lane] = runmax;
  __syncthreads();
  const float cm = cm_l[16 * w + (lane & 15)];

  float runsum = 0.f;
  stage(0, 0);
  __syncthreads();
  for (int qc = 0; qc < 32; ++qc) {
    const int buf = qc & 1;
    if (qc + 1 < 32) stage(qc + 1, buf ^ 1);
#pragma unroll
    for (int qt2 = 0; qt2 < 8; ++qt2) {
      v8f s = {};
#pragma unroll
      for (int kk = 0; kk < 2; ++kk) {
        v16bf a = frag_a(&qs_u[buf][0], 16 * qt2 + (lane & 15), 36, 16 * kk, lane);
        s = wmma_bf16(a, kb[kk], s);
      }
#pragma unroll
      for (int j = 0; j < 8; ++j) runsum += __expf(s[j] - cm);
    }
    __syncthreads();
  }
  runsum += __shfl_xor(runsum, 16, 32);
  if (lane < 16) {
    colmax[(size_t)b * N_ + k0 + 16 * w + lane] = runmax;
    colsum[(size_t)b * N_ + k0 + 16 * w + lane] = runsum;
  }
}

// ---------------- Kernel 3: O = softmax_cols(S) @ V ; out = gamma*O + x ----------
__global__ __launch_bounds__(256)
void attn_out_kernel(const unsigned short* __restrict__ Qbf,
                     const unsigned short* __restrict__ Kbf,
                     const unsigned short* __restrict__ Vbf,
                     const float* __restrict__ colmax, const float* __restrict__ colsum,
                     const float* __restrict__ x, const float* __restrict__ gamma,
                     float* __restrict__ out) {
  __shared__ unsigned int ks_u[2][64 * 36];                // K chunk [64 k][64 d], ping-pong
  __shared__ unsigned int vsT_u[2][256 * 33];              // V^T chunk [256 c][64 k], ping-pong
  __shared__ unsigned int ps_u[8 * 16 * 33];               // per-wave P [16 q][64 k]
  __shared__ float cm_l[2][64];
  __shared__ float rs_l[2][64];
  unsigned short* vsT_bf = (unsigned short*)vsT_u;
  unsigned short* ps_bf  = (unsigned short*)ps_u;

  const int b = blockIdx.x >> 5, qt = blockIdx.x & 31;
  const int q0 = qt * 128;
  const int tid = threadIdx.x, w = tid >> 5, lane = tid & 31;
  const int hi = (lane >= 16) ? 1 : 0;
  const unsigned int* Qu = (const unsigned int*)(Qbf + (size_t)b * N_ * D_);
  const unsigned int* Ku = (const unsigned int*)(Kbf + (size_t)b * N_ * D_);
  const unsigned int* Vu = (const unsigned int*)(Vbf + (size_t)b * N_ * C_);
  const uint4* Ku4 = reinterpret_cast<const uint4*>(Ku);
  const uint4* Vu4 = reinterpret_cast<const uint4*>(Vu);
  const float* cmg = colmax + (size_t)b * N_;
  const float* csg = colsum + (size_t)b * N_;

  v16bf qa[2];                                             // wave's Q rows, held
#pragma unroll
  for (int kk = 0; kk < 2; ++kk)
    qa[kk] = frag_a(Qu, q0 + 16 * w + (lane & 15), 32, 16 * kk, lane);

  v8f o[16];
  v8f zero = {};
#pragma unroll
  for (int t = 0; t < 16; ++t) o[t] = zero;

  auto stage = [&](int kc, int buf) {
    const int kbase = kc * 64;
    for (int idx = tid; idx < 64 * 8; idx += 256) {        // K: b128 in, b128 LDS out
      int r = idx >> 3, cu4 = idx & 7;
      uint4 val = Ku4[(size_t)(kbase + r) * 8 + cu4];
      *reinterpret_cast<uint4*>(&ks_u[buf][r * 36 + 4 * cu4]) = val;
    }
    for (int idx = tid; idx < 64 * 32; idx += 256) {       // V^T: b128 in, 16b transpose out
      int r = idx >> 5, c4 = idx & 31;
      uint4 val = Vu4[(size_t)(kbase + r) * 32 + c4];
      unsigned int uu[4] = {val.x, val.y, val.z, val.w};
      unsigned short* vb = vsT_bf + (size_t)buf * (256 * 66);
#pragma unroll
      for (int e = 0; e < 4; ++e) {
        int c = 8 * c4 + 2 * e;
        vb[c * 66 + r]       = (unsigned short)(uu[e] & 0xffffu);
        vb[(c + 1) * 66 + r] = (unsigned short)(uu[e] >> 16);
      }
    }
    if (tid < 64) {
      cm_l[buf][tid] = cmg[kbase + tid];
      rs_l[buf][tid] = 1.0f / csg[kbase + tid];
    }
  };

  stage(0, 0);
  __syncthreads();
  for (int kc = 0; kc < 64; ++kc) {
    const int buf = kc & 1;
    if (kc + 1 < 64) stage(kc + 1, buf ^ 1);               // overlap with WMMA below

    // S strip [16q x 64k], P = exp(S - colmax)/colsum -> LDS (wave-private)
#pragma unroll
    for (int ktile = 0; ktile < 4; ++ktile) {
      v8f s = {};
#pragma unroll
      for (int kk = 0; kk < 2; ++kk) {
        v16bf bfr = frag_b(&ks_u[buf][0], 16 * ktile + (lane & 15), 36, 16 * kk, lane);
        s = wmma_bf16(qa[kk], bfr, s);
      }
      int colk = 16 * ktile + (lane & 15);
      float cmv = cm_l[buf][colk], rsv = rs_l[buf][colk];
#pragma unroll
      for (int j = 0; j < 8; ++j) {
        float p = __expf(s[j] - cmv) * rsv;
        int m = j + 8 * hi;
        ps_bf[(w * 16 + m) * 66 + colk] = f2bf(p);
      }
    }
    __syncthreads();

    // O[16q x 256c] += P @ V
    const unsigned int* psw = ps_u + w * 16 * 33;
#pragma unroll
    for (int ct = 0; ct < 16; ++ct) {
#pragma unroll
      for (int kk = 0; kk < 2; ++kk) {
        v16bf a = frag_a(psw, (lane & 15), 33, 16 * kk, lane);
        v16bf bfr = frag_b(&vsT_u[buf][0], 16 * ct + (lane & 15), 33, 16 * kk, lane);
        o[ct] = wmma_bf16(a, bfr, o[ct]);
      }
    }
    __syncthreads();
  }

  const float g = gamma[0];
  const float* xb = x + (size_t)b * C_ * N_;
  float* ob = out + (size_t)b * C_ * N_;
#pragma unroll
  for (int ct = 0; ct < 16; ++ct) {
    int c = 16 * ct + (lane & 15);
    size_t base = (size_t)c * N_ + (size_t)(q0 + 16 * w + 8 * hi);
    float4 x0 = *reinterpret_cast<const float4*>(xb + base);
    float4 x1 = *reinterpret_cast<const float4*>(xb + base + 4);
    float4 r0, r1;
    r0.x = g * o[ct][0] + x0.x;  r0.y = g * o[ct][1] + x0.y;
    r0.z = g * o[ct][2] + x0.z;  r0.w = g * o[ct][3] + x0.w;
    r1.x = g * o[ct][4] + x1.x;  r1.y = g * o[ct][5] + x1.y;
    r1.z = g * o[ct][6] + x1.z;  r1.w = g * o[ct][7] + x1.w;
    *reinterpret_cast<float4*>(ob + base)     = r0;
    *reinterpret_cast<float4*>(ob + base + 4) = r1;
  }
}

extern "C" void kernel_launch(void* const* d_in, const int* in_sizes, int n_in,
                              void* d_out, int out_size, void* d_ws, size_t ws_size,
                              hipStream_t stream) {
  (void)in_sizes; (void)n_in; (void)out_size; (void)ws_size;
  const float* x     = (const float*)d_in[0];
  const float* Wq    = (const float*)d_in[1];
  const float* Wk    = (const float*)d_in[2];
  const float* Wv    = (const float*)d_in[3];
  const float* gamma = (const float*)d_in[4];
  float* out = (float*)d_out;

  char* ws = (char*)d_ws;
  unsigned short* Qbf = (unsigned short*)(ws);                       // 2 MB
  unsigned short* Kbf = (unsigned short*)(ws + (2ull << 20));        // 2 MB
  unsigned short* Vbf = (unsigned short*)(ws + (4ull << 20));        // 8 MB
  float* colmax = (float*)(ws + (12ull << 20));                      // 64 KB
  float* colsum = (float*)(ws + (12ull << 20) + (1ull << 18));       // 64 KB

  dim3 blk(256), grid(B_ * 32);
  qkv_kernel<<<grid, blk, 0, stream>>>(x, Wq, Qbf, 64);
  qkv_kernel<<<grid, blk, 0, stream>>>(x, Wk, Kbf, 64);
  qkv_kernel<<<grid, blk, 0, stream>>>(x, Wv, Vbf, 256);
  stats_kernel<<<grid, blk, 0, stream>>>(Qbf, Kbf, colmax, colsum);
  attn_out_kernel<<<grid, blk, 0, stream>>>(Qbf, Kbf, Vbf, colmax, colsum, x, gamma, out);
}
